// CAWN2_1786706395626
// MI455X (gfx1250) — matrix operand order repaired and checked by reading, not compiled, not run on hardware
//
#include <hip/hip_runtime.h>
#include <hip/hip_bf16.h>

// ---------------------------------------------------------------------------
// CAWN2 forward for MI455X (gfx1250, wave32, WMMA).
// Heavy GEMMs run on v_wmma_f32_16x16x32_bf16 (fp32 accumulate); gates,
// softmax, layernorm stay fp32 VALU. All staging through LDS.
// Grid order puts the N-tile dimension in blockIdx.x so all N-tiles of one
// M-tile are dispatch-adjacent -> activation re-reads hit the 192MB L2
// instead of HBM (avoids ~4GB of redundant HBM traffic in the GRU GEMM).
// ---------------------------------------------------------------------------

typedef __bf16 bf16_t;
typedef __attribute__((ext_vector_type(16))) __bf16 v16bf;
typedef __attribute__((ext_vector_type(8)))  float  v8f;

#define APAD 40  // LDS row pitch in bf16 (80B) -> conflict-free b128 frag loads

// fp32 -> bf16 round-to-nearest-even
__device__ __forceinline__ bf16_t f2bf(float f) {
  unsigned u = __builtin_bit_cast(unsigned, f);
  unsigned r = u + 0x7FFFu + ((u >> 16) & 1u);
  unsigned short h = (unsigned short)(r >> 16);
  return __builtin_bit_cast(bf16_t, h);
}

union FragU { v16bf v; uint4 q[2]; };

// A fragment (16x32 bf16, MxK). Per ISA 7.12.2: lane<16 -> M=lane, K in
// {0..7, 16..23}; lane>=16 -> M=lane-16, K in {8..15, 24..31}.
__device__ __forceinline__ v16bf frag_a(const bf16_t* tile, int rowbase, int lane) {
  int r  = rowbase + (lane & 15);
  int kh = (lane >> 4) * 8;
  const bf16_t* p = tile + r * APAD + kh;
  FragU f;
  f.q[0] = *(const uint4*)(p);       // K = kh .. kh+7
  f.q[1] = *(const uint4*)(p + 16);  // K = 16+kh .. 16+kh+7
  return f.v;
}

// B fragment (32x16 bf16, KxN), tile stored as [n][k] row-major (i.e. W rows).
// lanes 0-15: N=lane, K={0..7,16..23} ; lanes 16-31: N=lane-16, K={8..15,24..31}.
__device__ __forceinline__ v16bf frag_b(const bf16_t* tile, int rowbase, int lane) {
  int n   = rowbase + (lane & 15);
  int klo = (lane >> 4) * 8;
  const bf16_t* p = tile + n * APAD + klo;
  FragU f;
  f.q[0] = *(const uint4*)(p);
  f.q[1] = *(const uint4*)(p + 16);
  return f.v;
}

// ---------------------------------------------------------------------------
// fp32 -> bf16 convert
// ---------------------------------------------------------------------------
__global__ void cvt_f32_bf16(const float* __restrict__ s, bf16_t* __restrict__ d, int n) {
  int i = blockIdx.x * blockDim.x + threadIdx.x;
  if (i < n) d[i] = f2bf(s[i]);
}

// ---------------------------------------------------------------------------
// Fused GRU cell: seq = GRU(gru_x, gru_h) -> bf16 [65536 x 512]
// Block tile 64M x 32N, K=512.  6 WMMA accumulators per wave.
// Grid: (N/32, M/64)  -> N fast for L2 reuse of A tiles.
// ---------------------------------------------------------------------------
__global__ __launch_bounds__(256) void gru_wmma(
    const float* __restrict__ gx, const float* __restrict__ gh,
    const float* __restrict__ b_ih, const float* __restrict__ b_hh,
    const bf16_t* __restrict__ wih, const bf16_t* __restrict__ whh,
    bf16_t* __restrict__ seq) {
  __shared__ __align__(16) bf16_t Ax[64 * APAD];
  __shared__ __align__(16) bf16_t Ah[64 * APAD];
  __shared__ __align__(16) bf16_t Bs[6][32 * APAD];

  const int t = threadIdx.x;
  const int lane = t & 31;
  const int w = t >> 5, wm = w >> 1, wn = w & 1;
  const int n0 = blockIdx.x * 32;
  const int m0 = blockIdx.y * 64;

  v8f aXr = {0,0,0,0,0,0,0,0};
  v8f aXz = aXr, aXn = aXr, aHr = aXr, aHz = aXr, aHn = aXr;

  for (int k0 = 0; k0 < 512; k0 += 32) {
    __syncthreads();
    // stage 1280 chunks of 8 bf16: Ax(256) + Ah(256) + 6 W tiles(768)
#pragma unroll
    for (int i = 0; i < 5; ++i) {
      int c = t + i * 256;
      if (c < 512) {
        int isH = c >> 8;
        int cc = c & 255;
        int r = cc >> 2, cs = (cc & 3) * 8;
        const float* s = (isH ? gh : gx) + (size_t)(m0 + r) * 512 + k0 + cs;
        bf16_t* d = (isH ? Ah : Ax) + r * APAD + cs;
#pragma unroll
        for (int j = 0; j < 8; ++j) d[j] = f2bf(s[j]);
      } else {
        int cb = c - 512;
        int g = cb >> 7;                       // 0..5 : ih_r,ih_z,ih_n,hh_r,hh_z,hh_n
        int rr = (cb & 127) >> 2;
        int cs = (cb & 3) * 8;
        const bf16_t* wsrc = (g < 3) ? wih : whh;
        int grow = (g % 3) * 512 + n0 + rr;    // gate chunk row in W[1536x512]
        *(uint4*)&Bs[g][rr * APAD + cs] =
            *(const uint4*)(wsrc + (size_t)grow * 512 + k0 + cs);
      }
    }
    __syncthreads();

    v16bf ax = frag_a(Ax, wm * 16, lane);
    v16bf ah = frag_a(Ah, wm * 16, lane);
    v16bf b;
    b = frag_b(Bs[0], wn * 16, lane);
    aXr = __builtin_amdgcn_wmma_f32_16x16x32_bf16(false, ax, false, b, (short)0, aXr, false, false);
    b = frag_b(Bs[1], wn * 16, lane);
    aXz = __builtin_amdgcn_wmma_f32_16x16x32_bf16(false, ax, false, b, (short)0, aXz, false, false);
    b = frag_b(Bs[2], wn * 16, lane);
    aXn = __builtin_amdgcn_wmma_f32_16x16x32_bf16(false, ax, false, b, (short)0, aXn, false, false);
    b = frag_b(Bs[3], wn * 16, lane);
    aHr = __builtin_amdgcn_wmma_f32_16x16x32_bf16(false, ah, false, b, (short)0, aHr, false, false);
    b = frag_b(Bs[4], wn * 16, lane);
    aHz = __builtin_amdgcn_wmma_f32_16x16x32_bf16(false, ah, false, b, (short)0, aHz, false, false);
    b = frag_b(Bs[5], wn * 16, lane);
    aHn = __builtin_amdgcn_wmma_f32_16x16x32_bf16(false, ah, false, b, (short)0, aHn, false, false);
  }

  // C layout: lane<16 -> M=j, N=lane ; lane>=16 -> M=8+j, N=lane-16
  int nloc = lane & 15;
  int mofs = (lane >> 4) * 8;
  int n = n0 + wn * 16 + nloc;
  float bir = b_ih[n],        bhr = b_hh[n];
  float biz = b_ih[512 + n],  bhz = b_hh[512 + n];
  float bin_ = b_ih[1024 + n], bhn = b_hh[1024 + n];
#pragma unroll
  for (int j = 0; j < 8; ++j) {
    int m = m0 + wm * 16 + mofs + j;
    float r = 1.0f / (1.0f + __expf(-(aXr[j] + bir + aHr[j] + bhr)));
    float z = 1.0f / (1.0f + __expf(-(aXz[j] + biz + aHz[j] + bhz)));
    float nn = tanhf(aXn[j] + bin_ + r * (aHn[j] + bhn));
    float hp = gh[(size_t)m * 512 + n];
    seq[(size_t)m * 512 + n] = f2bf((1.0f - z) * nn + z * hp);
  }
}

// ---------------------------------------------------------------------------
// Generic bf16 WMMA GEMM:  C[MxN] = act(A[MxK] @ W[NxK]^T + bias)
// Block tile 64M x 64N, 8 waves (4x2), 2 accumulators per wave (A-frag
// reused for 2 WMMAs).  Grid: (N/64, M/64)  -> N fast for L2 reuse.
// A may be fp32 (converted while staging) or bf16; bf16 A may be a concat of
// two row-major halves split at `ksplit` (lda of A == ksplit).
// ---------------------------------------------------------------------------
__global__ __launch_bounds__(256) void gemm_bf16_wmma(
    const void* __restrict__ A, int a_f32, const void* __restrict__ A2, int ksplit,
    const bf16_t* __restrict__ W, const float* __restrict__ bias,
    float* __restrict__ Cf, bf16_t* __restrict__ Cb, int relu,
    int N, int K) {
  __shared__ __align__(16) bf16_t As[64 * APAD];
  __shared__ __align__(16) bf16_t Bs[64 * APAD];
  const int t = threadIdx.x;
  const int lane = t & 31;
  const int w = t >> 5, wm = w >> 1, wn = w & 1;
  const int n0 = blockIdx.x * 64;
  const int m0 = blockIdx.y * 64;

  v8f acc0 = {0,0,0,0,0,0,0,0};
  v8f acc1 = acc0;
  for (int k0 = 0; k0 < K; k0 += 32) {
    __syncthreads();
    {
      int r = t >> 2, cs = (t & 3) * 8;
      int kk = k0 + cs;
      int row = m0 + r;
      bf16_t* d = As + r * APAD + cs;
      if (a_f32) {
        const float* s = (const float*)A + (size_t)row * K + kk;
#pragma unroll
        for (int j = 0; j < 8; ++j) d[j] = f2bf(s[j]);
      } else if (kk < ksplit) {
        *(uint4*)d = *(const uint4*)((const bf16_t*)A + (size_t)row * ksplit + kk);
      } else {
        *(uint4*)d = *(const uint4*)((const bf16_t*)A2 + (size_t)row * (K - ksplit) + (kk - ksplit));
      }
      // B tile: 64 rows x 32 cols -> 256 chunks, one per thread
      *(uint4*)(Bs + r * APAD + cs) =
          *(const uint4*)(W + (size_t)(n0 + r) * K + k0 + cs);
    }
    __syncthreads();
    v16bf af = frag_a(As, wm * 16, lane);
    v16bf b0 = frag_b(Bs, wn * 32, lane);
    acc0 = __builtin_amdgcn_wmma_f32_16x16x32_bf16(false, af, false, b0, (short)0, acc0, false, false);
    v16bf b1 = frag_b(Bs, wn * 32 + 16, lane);
    acc1 = __builtin_amdgcn_wmma_f32_16x16x32_bf16(false, af, false, b1, (short)0, acc1, false, false);
  }

  int nloc = lane & 15;
  int mofs = (lane >> 4) * 8;
#pragma unroll
  for (int sub = 0; sub < 2; ++sub) {
    const v8f& acc = sub ? acc1 : acc0;
    int n = n0 + wn * 32 + sub * 16 + nloc;
    float bv = bias ? bias[n] : 0.0f;
#pragma unroll
    for (int j = 0; j < 8; ++j) {
      int m = m0 + wm * 16 + mofs + j;
      float v = acc[j] + bv;
      if (relu) v = fmaxf(v, 0.0f);
      if (Cf) Cf[(size_t)m * N + n] = v;
      if (Cb) Cb[(size_t)m * N + n] = f2bf(v);
    }
  }
}

// ---------------------------------------------------------------------------
// Attention per (b, s): scores (H=4 x NGH=64, DK=128), mask, softmax,
// attn @ v.  attn probs written straight into d_out tail.
// ---------------------------------------------------------------------------
__global__ __launch_bounds__(256) void attn_kernel(
    const float* __restrict__ qproj, const bf16_t* __restrict__ kproj,
    const bf16_t* __restrict__ vproj, const int* __restrict__ mask,
    float* __restrict__ attn_out, float* __restrict__ attn_probs) {
  const int bs = blockIdx.x;  // b*16 + s
  const int t = threadIdx.x;
  __shared__ float qs[512];
  __shared__ float sc[256];   // [h][n] = [4][64]

  qs[t] = qproj[(size_t)bs * 512 + t];
  qs[t + 256] = qproj[(size_t)bs * 512 + t + 256];
  __syncthreads();

  int h = t >> 6, n = t & 63;
  size_t krow = ((size_t)bs * 64 + n) * 512 + (size_t)h * 128;
  float s = 0.0f;
#pragma unroll 8
  for (int d = 0; d < 128; ++d) s += qs[h * 128 + d] * (float)kproj[krow + d];
  s *= 0.08838834764831845f;  // 1/sqrt(128)
  int b = bs >> 4;
  if (mask[b * 1024 + (bs & 15) * 64 + n]) s = -1e10f;
  sc[t] = s;
  __syncthreads();

  if (t < 4) {  // softmax over the 64 neighbors of head t
    float mx = -3.0e38f;
    for (int i = 0; i < 64; ++i) mx = fmaxf(mx, sc[t * 64 + i]);
    float sum = 0.0f;
    for (int i = 0; i < 64; ++i) { float e = __expf(sc[t * 64 + i] - mx); sc[t * 64 + i] = e; sum += e; }
    float inv = 1.0f / sum;
    for (int i = 0; i < 64; ++i) sc[t * 64 + i] *= inv;
  }
  __syncthreads();

  attn_probs[(size_t)bs * 256 + t] = sc[t];

#pragma unroll
  for (int rep = 0; rep < 2; ++rep) {
    int e = t + rep * 256;      // output feature 0..511
    int hh = e >> 7;
    float o = 0.0f;
    size_t vbase = (size_t)bs * 64 * 512 + e;
    for (int n2 = 0; n2 < 64; ++n2)
      o += sc[hh * 64 + n2] * (float)vproj[vbase + (size_t)n2 * 512];
    attn_out[(size_t)bs * 512 + e] = o;
  }
}

// ---------------------------------------------------------------------------
// LayerNorm(wo_out + src) * gamma + beta -> bf16 (row = one (b,s))
// ---------------------------------------------------------------------------
__global__ __launch_bounds__(256) void ln_res_kernel(
    const float* __restrict__ wo_out, const float* __restrict__ src,
    const float* __restrict__ gamma, const float* __restrict__ beta,
    bf16_t* __restrict__ out2) {
  const int row = blockIdx.x;
  const int t = threadIdx.x;
  __shared__ float red[256];
  float x0 = wo_out[(size_t)row * 512 + t]       + src[(size_t)row * 512 + t];
  float x1 = wo_out[(size_t)row * 512 + t + 256] + src[(size_t)row * 512 + t + 256];
  red[t] = x0 + x1;
  __syncthreads();
  for (int o = 128; o > 0; o >>= 1) { if (t < o) red[t] += red[t + o]; __syncthreads(); }
  float mu = red[0] * (1.0f / 512.0f);
  __syncthreads();
  float d0 = x0 - mu, d1 = x1 - mu;
  red[t] = d0 * d0 + d1 * d1;
  __syncthreads();
  for (int o = 128; o > 0; o >>= 1) { if (t < o) red[t] += red[t + o]; __syncthreads(); }
  float rstd = rsqrtf(red[0] * (1.0f / 512.0f) + 1e-5f);
  out2[(size_t)row * 512 + t]       = f2bf(d0 * rstd * gamma[t] + beta[t]);
  out2[(size_t)row * 512 + t + 256] = f2bf(d1 * rstd * gamma[t + 256] + beta[t + 256]);
}

// ---------------------------------------------------------------------------
extern "C" void kernel_launch(void* const* d_in, const int* in_sizes, int n_in,
                              void* d_out, int out_size, void* d_ws, size_t ws_size,
                              hipStream_t stream) {
  (void)in_sizes; (void)n_in; (void)out_size; (void)ws_size;
  const float* src   = (const float*)d_in[0];
  const float* gru_x = (const float*)d_in[1];
  const float* gru_h = (const float*)d_in[2];
  const int*   mask  = (const int*)d_in[3];
  const float* W_ih  = (const float*)d_in[4];
  const float* W_hh  = (const float*)d_in[5];
  const float* b_ih  = (const float*)d_in[6];
  const float* b_hh  = (const float*)d_in[7];
  const float* Wq    = (const float*)d_in[8];
  const float* Wk    = (const float*)d_in[9];
  const float* Wv    = (const float*)d_in[10];
  const float* Wo    = (const float*)d_in[11];
  const float* bo    = (const float*)d_in[12];
  const float* gamma = (const float*)d_in[13];
  const float* beta  = (const float*)d_in[14];
  const float* W1    = (const float*)d_in[15];
  const float* b1    = (const float*)d_in[16];
  const float* W2    = (const float*)d_in[17];
  const float* b2    = (const float*)d_in[18];

  char* ws = (char*)d_ws;
  size_t off = 0;
  auto alloc = [&](size_t bytes) -> void* {
    void* p = ws + off; off += (bytes + 255) & ~(size_t)255; return p;
  };
  bf16_t* wihb   = (bf16_t*)alloc(1536 * 512 * 2);
  bf16_t* whhb   = (bf16_t*)alloc(1536 * 512 * 2);
  bf16_t* wqb    = (bf16_t*)alloc(512 * 512 * 2);
  bf16_t* wkb    = (bf16_t*)alloc(512 * 512 * 2);
  bf16_t* wvb    = (bf16_t*)alloc(512 * 512 * 2);
  bf16_t* wob    = (bf16_t*)alloc(512 * 512 * 2);
  bf16_t* w1b    = (bf16_t*)alloc(512 * 1024 * 2);
  bf16_t* w2b    = (bf16_t*)alloc(512 * 512 * 2);
  bf16_t* srcb   = (bf16_t*)alloc((size_t)1024 * 512 * 2);
  bf16_t* seqb   = (bf16_t*)alloc((size_t)65536 * 512 * 2);
  bf16_t* kprojb = (bf16_t*)alloc((size_t)65536 * 512 * 2);
  bf16_t* vprojb = (bf16_t*)alloc((size_t)65536 * 512 * 2);
  float*  qproj  = (float*)alloc((size_t)1024 * 512 * 4);
  float*  attn_o = (float*)alloc((size_t)1024 * 512 * 4);
  float*  wo_out = (float*)alloc((size_t)1024 * 512 * 4);
  bf16_t* out2b  = (bf16_t*)alloc((size_t)1024 * 512 * 2);
  bf16_t* hmrgb  = (bf16_t*)alloc((size_t)1024 * 512 * 2);

  auto cvt = [&](const float* s, bf16_t* d, int n) {
    cvt_f32_bf16<<<dim3((n + 255) / 256), dim3(256), 0, stream>>>(s, d, n);
  };
  cvt(W_ih, wihb, 1536 * 512);
  cvt(W_hh, whhb, 1536 * 512);
  cvt(Wq, wqb, 512 * 512);
  cvt(Wk, wkb, 512 * 512);
  cvt(Wv, wvb, 512 * 512);
  cvt(Wo, wob, 512 * 512);
  cvt(W1, w1b, 512 * 1024);
  cvt(W2, w2b, 512 * 512);
  cvt(src, srcb, 1024 * 512);

  // GRU: seq = GRUCell(gru_x, gru_h)           [65536 x 512] bf16
  gru_wmma<<<dim3(16, 1024), dim3(256), 0, stream>>>(
      gru_x, gru_h, b_ih, b_hh, wihb, whhb, seqb);

  // K / V projections: seq @ Wk^T, seq @ Wv^T  [65536 x 512] bf16
  gemm_bf16_wmma<<<dim3(8, 1024), dim3(256), 0, stream>>>(
      seqb, 0, nullptr, 512, wkb, nullptr, nullptr, kprojb, 0, 512, 512);
  gemm_bf16_wmma<<<dim3(8, 1024), dim3(256), 0, stream>>>(
      seqb, 0, nullptr, 512, wvb, nullptr, nullptr, vprojb, 0, 512, 512);

  // Q projection: src @ Wq^T                   [1024 x 512] fp32
  gemm_bf16_wmma<<<dim3(8, 16), dim3(256), 0, stream>>>(
      srcb, 0, nullptr, 512, wqb, nullptr, qproj, nullptr, 0, 512, 512);

  // Attention (scores + softmax + attn@v); probs -> d_out tail
  float* attn_probs = (float*)d_out + 524288;  // after z_out [64*16*512]
  attn_kernel<<<dim3(1024), dim3(256), 0, stream>>>(
      qproj, kprojb, vprojb, mask, attn_o, attn_probs);

  // Output projection: attn_o @ Wo^T + bo      [1024 x 512] fp32
  gemm_bf16_wmma<<<dim3(8, 16), dim3(256), 0, stream>>>(
      attn_o, 1, nullptr, 512, wob, bo, wo_out, nullptr, 0, 512, 512);

  // LayerNorm(wo_out + src)                    [1024 x 512] bf16
  ln_res_kernel<<<dim3(1024), dim3(256), 0, stream>>>(
      wo_out, src, gamma, beta, out2b);

  // Merge 1: relu([out2 | src] @ W1^T + b1)    K=1024 concat, bf16 out
  gemm_bf16_wmma<<<dim3(8, 16), dim3(256), 0, stream>>>(
      out2b, 0, srcb, 512, w1b, b1, nullptr, hmrgb, 1, 512, 1024);

  // Merge 2: hmerge @ W2^T + b2 -> z_out (d_out head, fp32)
  gemm_bf16_wmma<<<dim3(8, 16), dim3(256), 0, stream>>>(
      hmrgb, 0, nullptr, 512, w2b, b2, (float*)d_out, nullptr, 0, 512, 512);
}
